// ObjectMsgEncoder_31817117728824
// MI455X (gfx1250) — compile-verified, matrix operand order checked by reading, not run on hardware
//
#include <hip/hip_runtime.h>
#include <math.h>

// ---------------- problem constants ----------------
constexpr int NB = 256;                 // batch
constexpr int NS = 28;                  // objects per scene
constexpr int NO = 256;                 // lstm/out dim
constexpr int NG4 = 4 * NO;             // 1024 (lstm gates)
constexpr int NBS = NB * NS;            // 7168
constexpr long long NBSS = (long long)NB * NS * NS; // 200704

typedef __attribute__((ext_vector_type(16))) __bf16 v16bf;
typedef __attribute__((ext_vector_type(4)))  __bf16 v4bf;
typedef __attribute__((ext_vector_type(8)))  float  v8f;

// native f32 -> bf16 (RNE); clang lowers to v_cvt bf16 ops on gfx1250
__device__ inline __bf16 f2bf(float f) { return (__bf16)f; }

// pack 4 floats -> 4 bf16, one 8-byte LDS store
__device__ inline void store4bf(__bf16* dst, float4 v) {
  v4bf h;
  h.x = (__bf16)v.x;
  h.y = (__bf16)v.y;
  h.z = (__bf16)v.z;
  h.w = (__bf16)v.w;
  *(v4bf*)dst = h;
}
__device__ inline float sigmf(float x) { return 1.0f / (1.0f + __expf(-x)); }

// One lane's 16 bf16 elements of a WMMA A(row)/B(col) fragment.
// `rowp` points at the 32 contiguous K values of this lane's row/column.
// Lanes 0-15 hold K={0..7,16..23}; lanes 16-31 hold K={8..15,24..31}.
__device__ inline v16bf load_frag(const __bf16* rowp, int hi) {
  int koff = hi << 3;
  v16bf r;
#pragma unroll
  for (int e = 0; e < 8; ++e) r[e] = rowp[koff + e];
#pragma unroll
  for (int e = 0; e < 8; ++e) r[8 + e] = rowp[16 + koff + e];
  return r;
}

// ---------------- generic GEMM:  C[M,N] = A[M,K] @ W[N,K]^T (+bias[n]) (+add[m,n]) ----
// A: f32 row-major (lda, multiple of 4). W: bf16 row-major [N,K].
// Requires M % 32 == 0, N % 64 == 0, K % 32 == 0. 256 threads, tile 32x64.
__global__ void gemm_bf16_kernel(const float* __restrict__ A, int lda,
                                 const __bf16* __restrict__ W, int K,
                                 const float* __restrict__ bias,
                                 const float* __restrict__ add, long long ldadd,
                                 float* __restrict__ C, int ldc, int M) {
  __shared__ __bf16 As[32][32];
  __shared__ __bf16 Bs[64][32];
  const int tid = threadIdx.x, lane = tid & 31, wave = tid >> 5;
  const int hi = lane >> 4, lo = lane & 15;
  const int mBase = blockIdx.x * 32;
  const int nBase = blockIdx.y * 64;
  const int rt = wave & 1, ct = wave >> 1;
  // A-tile chunk: one float4 per thread
  const int ar = tid >> 3, ac = (tid & 7) * 4;
  // W-tile chunk: one uint4 (8 bf16) per thread
  const int wn = tid >> 2, wc = (tid & 3) * 8;
  v8f acc = {};
  for (int k0 = 0; k0 < K; k0 += 32) {
    __syncthreads();
    {
      float4 v = *(const float4*)(A + (size_t)(mBase + ar) * lda + k0 + ac);
      store4bf(&As[ar][ac], v);
      *(uint4*)&Bs[wn][wc] =
          *(const uint4*)(W + (size_t)(nBase + wn) * K + k0 + wc);
    }
    __syncthreads();
    v16bf af = load_frag(&As[rt * 16 + lo][0], hi);
    v16bf bf = load_frag(&Bs[ct * 16 + lo][0], hi);
    acc = __builtin_amdgcn_wmma_f32_16x16x32_bf16(false, af, false, bf,
                                                  (short)0, acc, false, false);
  }
  const int col = nBase + ct * 16 + lo;
  const float bv = bias ? bias[col] : 0.0f;
#pragma unroll
  for (int v = 0; v < 8; ++v) {
    int row = mBase + rt * 16 + v + (hi << 3);
    float val = acc[v] + bv;
    if (add) val += add[(size_t)row * ldadd + col];
    C[(size_t)row * ldc + col] = val;
  }
}

// ---------------- fused relation layer ----------------
// For 64 rows (b,i,j) of rlf (512 threads = 16 waves, 4x16 wmma tiles):
//   lin = rlf@W3^T + P1[b,i] + P2[b,j] + bgr            (phase 1, WMMA)
//   cp  = lin@Wbc^T + bbc                               (phase 2, WMMA)
//   rlf = tanh(s0*lin + s1*cp) * rel_mask               (in place)
__global__ void layer_fused_kernel(float* __restrict__ RLF,
                                   const float* __restrict__ P1,
                                   const float* __restrict__ P2,
                                   const __bf16* __restrict__ W3,
                                   const float* __restrict__ bgr,
                                   const __bf16* __restrict__ WBC,
                                   const float* __restrict__ bbc,
                                   const float* __restrict__ alpha,
                                   const int* __restrict__ counts) {
  __shared__ __bf16 As[64][32];     //  4 KB
  __shared__ __bf16 Bs[256][32];    // 16 KB
  __shared__ __bf16 Cb[64][256];    // 32 KB
  const int tid = threadIdx.x, lane = tid & 31, wave = tid >> 5;
  const int hi = lane >> 4, lo = lane & 15;
  const size_t mBase = (size_t)blockIdx.x * 64;
  const int rt = wave & 3;                  // row tile 0..3
  const int ctb = (wave >> 2) * 4;          // 4 column tiles per wave
  // A-tile chunk: one float4 per thread (64*8 = 512 chunks)
  const int ar = tid >> 3, ac = (tid & 7) * 4;
  // W-tile chunks: two uint4 per thread (256*4 = 1024 chunks)
  const int wn0 = tid >> 2, wc0 = (tid & 3) * 8;

  // per-wave column constants
  float bgrv[4], bbcv[4];
#pragma unroll
  for (int q = 0; q < 4; ++q) {
    int col = (ctb + q) * 16 + lo;
    bgrv[q] = bgr[col];
    bbcv[q] = bbc[col];
  }

  v8f acc[4] = {};
  // -------- phase 1: rlf @ W3^T --------
  for (int k0 = 0; k0 < 256; k0 += 32) {
    __syncthreads();
    {
      float4 v = *(const float4*)(RLF + (mBase + ar) * 256 + k0 + ac);
      store4bf(&As[ar][ac], v);
      *(uint4*)&Bs[wn0][wc0] =
          *(const uint4*)(W3 + (size_t)wn0 * 256 + k0 + wc0);
      *(uint4*)&Bs[wn0 + 128][wc0] =
          *(const uint4*)(W3 + (size_t)(wn0 + 128) * 256 + k0 + wc0);
    }
    __syncthreads();
    v16bf af = load_frag(&As[rt * 16 + lo][0], hi);
#pragma unroll
    for (int q = 0; q < 4; ++q) {
      v16bf bf = load_frag(&Bs[(ctb + q) * 16 + lo][0], hi);
      acc[q] = __builtin_amdgcn_wmma_f32_16x16x32_bf16(false, af, false, bf,
                                                       (short)0, acc[q], false, false);
    }
  }
  // add P1[b,i] + P2[b,j] + bgr; keep f32 lin in registers + bf16 copy in LDS
  float flin[4][8];
#pragma unroll
  for (int v = 0; v < 8; ++v) {
    int rl = rt * 16 + v + (hi << 3);
    size_t rg = mBase + rl;
    int j = (int)(rg % NS);
    int i = (int)((rg / NS) % NS);
    int b = (int)(rg / ((size_t)NS * NS));
    const float* p1r = P1 + ((size_t)b * NS + i) * 256;
    const float* p2r = P2 + ((size_t)b * NS + j) * 256;
#pragma unroll
    for (int q = 0; q < 4; ++q) {
      int col = (ctb + q) * 16 + lo;
      float val = acc[q][v] + p1r[col] + p2r[col] + bgrv[q];
      flin[q][v] = val;
      Cb[rl][col] = f2bf(val);
    }
  }
  // -------- phase 2: lin @ Wbc^T --------
  v8f acc2[4] = {};
  for (int k0 = 0; k0 < 256; k0 += 32) {
    __syncthreads();
    {
      *(uint4*)&Bs[wn0][wc0] =
          *(const uint4*)(WBC + (size_t)wn0 * 256 + k0 + wc0);
      *(uint4*)&Bs[wn0 + 128][wc0] =
          *(const uint4*)(WBC + (size_t)(wn0 + 128) * 256 + k0 + wc0);
    }
    __syncthreads();
    v16bf af = load_frag(&Cb[rt * 16 + lo][k0], hi);
#pragma unroll
    for (int q = 0; q < 4; ++q) {
      v16bf bf = load_frag(&Bs[(ctb + q) * 16 + lo][0], hi);
      acc2[q] = __builtin_amdgcn_wmma_f32_16x16x32_bf16(false, af, false, bf,
                                                        (short)0, acc2[q], false, false);
    }
  }
  // -------- bezier + tanh + mask, write back --------
  const float t = sigmf(alpha[0]);
  const float s0 = (1.0f - t) * (1.0f - t) + t * t;
  const float s1 = 2.0f * (1.0f - t) * t;
#pragma unroll
  for (int v = 0; v < 8; ++v) {
    int rl = rt * 16 + v + (hi << 3);
    size_t rg = mBase + rl;
    int j = (int)(rg % NS);
    int i = (int)((rg / NS) % NS);
    int b = (int)(rg / ((size_t)NS * NS));
    int cnt = counts[b];
    float mask = (i < cnt && j < cnt) ? 1.0f : 0.0f;
#pragma unroll
    for (int q = 0; q < 4; ++q) {
      int col = (ctb + q) * 16 + lo;
      float bez = s0 * flin[q][v] + s1 * (acc2[q][v] + bbcv[q]);
      RLF[rg * 256 + col] = tanhf(bez) * mask;
    }
  }
}

// ---------------- misc elementwise kernels ----------------
__global__ void zero_kernel(float* __restrict__ p, int n) {
  int i = blockIdx.x * blockDim.x + threadIdx.x;
  if (i < n) p[i] = 0.0f;
}

__global__ void olf_init_kernel(const float* __restrict__ emb,
                                const int* __restrict__ counts,
                                float* __restrict__ OLF) {
  size_t idx = (size_t)blockIdx.x * blockDim.x + threadIdx.x;
  if (idx >= (size_t)NBS * 256) return;
  size_t row = idx >> 8;
  int s = (int)(row % NS), b = (int)(row / NS);
  float m = (s < counts[b]) ? 1.0f : 0.0f;
  OLF[idx] = emb[idx] * m;
}

// rlf0[b,i,j,e] = sum_d (cp_i[d]-cp_j[d]) * Wr[e,d] + br[e]   (K=3: plain VALU)
__global__ void rlf0_kernel(const float* __restrict__ centers,
                            const int* __restrict__ counts,
                            const float* __restrict__ Wr,
                            const float* __restrict__ br,
                            float* __restrict__ RLF) {
  int e = threadIdx.x;
  size_t r = blockIdx.x;
  int j = (int)(r % NS), i = (int)((r / NS) % NS), b = (int)(r / ((size_t)NS * NS));
  int cnt = counts[b];
  float mi = (i < cnt) ? 1.0f : 0.0f;
  float mj = (j < cnt) ? 1.0f : 0.0f;
  const float* ci = centers + ((size_t)b * NS + i) * 3;
  const float* cj = centers + ((size_t)b * NS + j) * 3;
  float acc = br[e];
#pragma unroll
  for (int d = 0; d < 3; ++d)
    acc += (ci[d] * mi - cj[d] * mj) * Wr[e * 3 + d];
  RLF[r * 256 + e] = acc;
}

// LSTM gate math for one timestep (PyTorch gate order i,f,g,o)
__global__ void lstm_gate_kernel(const float* __restrict__ Z,
                                 float* __restrict__ H, float* __restrict__ Cst,
                                 float* __restrict__ OLF2, int t,
                                 const int* __restrict__ counts) {
  int idx = blockIdx.x * blockDim.x + threadIdx.x;
  if (idx >= NB * NO) return;
  int b = idx >> 8, o = idx & 255;
  const float* z = Z + (size_t)b * NG4;
  float ig = sigmf(z[o]);
  float fg = sigmf(z[o + 256]);
  float gg = tanhf(z[o + 512]);
  float og = sigmf(z[o + 768]);
  float c = fg * Cst[idx] + ig * gg;
  float h = og * tanhf(c);
  Cst[idx] = c;
  H[idx] = h;
  float m = (t < counts[b]) ? 1.0f : 0.0f;
  OLF2[((size_t)b * NS + t) * 256 + o] = h * m;
}

// gscore[row] = mean_d (R2[row]·Wmu[d] + bmu[d]) = R2[row]·mean(Wmu) + mean(bmu)
__global__ void gscore_kernel(const float* __restrict__ R2,
                              const float* __restrict__ Wmu,
                              const float* __restrict__ bmu,
                              float* __restrict__ GS) {
  int lane = threadIdx.x & 31, wave = threadIdx.x >> 5;
  size_t row = (size_t)blockIdx.x * 8 + wave;
  if (row >= (size_t)NBSS) return;
  const float* x = R2 + row * 256;
  float acc = 0.0f;
  for (int o = lane; o < 256; o += 32) {
    float w = (Wmu[o] + Wmu[256 + o] + Wmu[512 + o]) * (1.0f / 3.0f);
    acc += x[o] * w;
  }
  for (int off = 16; off > 0; off >>= 1) acc += __shfl_down(acc, off, 32);
  if (lane == 0) GS[row] = acc + (bmu[0] + bmu[1] + bmu[2]) * (1.0f / 3.0f);
}

// softmax over axis=1 (index i): per (b,j,o) max & sum-exp; per (b,j) gaussian stats
__global__ void softmax_stats_kernel(const float* __restrict__ R2,
                                     const float* __restrict__ GS,
                                     float* __restrict__ BM, float* __restrict__ SMS,
                                     float* __restrict__ GM, float* __restrict__ GSUM) {
  int o = threadIdx.x;
  int bj = blockIdx.x;                 // b*NS + j
  int b = bj / NS, j = bj % NS;
  float m = -1e30f;
  for (int i = 0; i < NS; ++i)
    m = fmaxf(m, R2[(((size_t)b * NS + i) * NS + j) * 256 + o]);
  float s = 0.0f;
  for (int i = 0; i < NS; ++i)
    s += __expf(R2[(((size_t)b * NS + i) * NS + j) * 256 + o] - m);
  BM[(size_t)bj * 256 + o] = m;
  SMS[(size_t)bj * 256 + o] = s;
  if (o == 0) {
    float gm = -1e30f;
    for (int i = 0; i < NS; ++i)
      gm = fmaxf(gm, GS[((size_t)b * NS + i) * NS + j]);
    float gs = 0.0f;
    for (int i = 0; i < NS; ++i)
      gs += __expf(GS[((size_t)b * NS + i) * NS + j] - gm);
    GM[bj] = gm;
    GSUM[bj] = gs;
  }
}

// rel_out[b,i,o] = sum_j (0.5*battn + 0.5*gattn) * rlf2[b,i,j,o] * obj_mask[b,i]
__global__ void attn_out_kernel(const float* __restrict__ R2,
                                const float* __restrict__ GS,
                                const float* __restrict__ BM, const float* __restrict__ SMS,
                                const float* __restrict__ GM, const float* __restrict__ GSUM,
                                const int* __restrict__ counts,
                                float* __restrict__ OUT) {
  int o = threadIdx.x;
  int bi = blockIdx.x;                 // b*NS + i
  int b = bi / NS, i = bi % NS;
  float acc = 0.0f;
  size_t base = (size_t)bi * NS;
  for (int j = 0; j < NS; ++j) {
    size_t row = base + j;
    float x = R2[row * 256 + o];
    int bj = b * NS + j;
    float ba = __expf(x - BM[(size_t)bj * 256 + o]) / SMS[(size_t)bj * 256 + o];
    float ga = __expf(GS[row] - GM[bj]) / GSUM[bj];
    acc += (0.5f * ba + 0.5f * ga) * x;
  }
  float m = (i < counts[b]) ? 1.0f : 0.0f;
  OUT[(size_t)bi * 256 + o] = acc * m;
}

// weight prep: f32 -> bf16 (+ Wgr split, bias fold)
__global__ void prep_kernel(const float* __restrict__ W_ih, const float* __restrict__ W_hh,
                            const float* __restrict__ Wgr, const float* __restrict__ Wbc,
                            const float* __restrict__ Wesa,
                            const float* __restrict__ b_ih, const float* __restrict__ b_hh,
                            __bf16* WIH, __bf16* WHH, __bf16* W1, __bf16* W2, __bf16* W3,
                            __bf16* WBCb, __bf16* WESb, float* BIAS4) {
  const size_t NIH = (size_t)NG4 * 256;   // 262144
  const size_t NSQ = 256 * 256;           // 65536
  const size_t total = 2 * NIH + 5 * NSQ + NG4;
  size_t stride = (size_t)gridDim.x * blockDim.x;
  for (size_t t0 = (size_t)blockIdx.x * blockDim.x + threadIdx.x; t0 < total; t0 += stride) {
    size_t t = t0;
    if (t < NIH) { WIH[t] = f2bf(W_ih[t]); continue; } t -= NIH;
    if (t < NIH) { WHH[t] = f2bf(W_hh[t]); continue; } t -= NIH;
    if (t < NSQ) { size_t o = t >> 8, k = t & 255; W1[t] = f2bf(Wgr[o * 768 + k]); continue; } t -= NSQ;
    if (t < NSQ) { size_t o = t >> 8, k = t & 255; W2[t] = f2bf(Wgr[o * 768 + 256 + k]); continue; } t -= NSQ;
    if (t < NSQ) { size_t o = t >> 8, k = t & 255; W3[t] = f2bf(Wgr[o * 768 + 512 + k]); continue; } t -= NSQ;
    if (t < NSQ) { WBCb[t] = f2bf(Wbc[t]); continue; } t -= NSQ;
    if (t < NSQ) { WESb[t] = f2bf(Wesa[t]); continue; } t -= NSQ;
    BIAS4[t] = b_ih[t] + b_hh[t];
  }
}

// ---------------- host-side orchestration ----------------
extern "C" void kernel_launch(void* const* d_in, const int* in_sizes, int n_in,
                              void* d_out, int out_size, void* d_ws, size_t ws_size,
                              hipStream_t stream) {
  (void)in_sizes; (void)n_in; (void)out_size; (void)ws_size;
  const float* centers = (const float*)d_in[0];
  const float* emb     = (const float*)d_in[1];
  const int*   counts  = (const int*)d_in[2];
  const float* Wr      = (const float*)d_in[3];
  const float* br      = (const float*)d_in[4];
  const float* W_ih    = (const float*)d_in[5];
  const float* W_hh    = (const float*)d_in[6];
  const float* b_ih    = (const float*)d_in[7];
  const float* b_hh    = (const float*)d_in[8];
  const float* Wgr     = (const float*)d_in[9];
  const float* bgr     = (const float*)d_in[10];
  const float* Wbc     = (const float*)d_in[11];
  const float* bbc     = (const float*)d_in[12];
  const float* alpha   = (const float*)d_in[13];
  const float* Wesa    = (const float*)d_in[14];
  const float* besa    = (const float*)d_in[15];
  const float* Wmu     = (const float*)d_in[16];
  const float* bmu     = (const float*)d_in[17];
  // d_in[18], d_in[19] (Wlv, blv): logvars unused at eval — skipped.

  char* w = (char*)d_ws;
  size_t off = 0;
  auto alloc = [&](size_t bytes) -> void* {
    void* p = w + off;
    off = (off + bytes + 255) & ~(size_t)255;
    return p;
  };
  float*  RLF   = (float*)alloc((size_t)NBSS * 256 * 4);
  float*  R2    = (float*)alloc((size_t)NBSS * 256 * 4);
  __bf16* WIH   = (__bf16*)alloc((size_t)NG4 * 256 * 2);
  __bf16* WHH   = (__bf16*)alloc((size_t)NG4 * 256 * 2);
  __bf16* W1    = (__bf16*)alloc(256 * 256 * 2);
  __bf16* W2    = (__bf16*)alloc(256 * 256 * 2);
  __bf16* W3    = (__bf16*)alloc(256 * 256 * 2);
  __bf16* WBCb  = (__bf16*)alloc(256 * 256 * 2);
  __bf16* WESb  = (__bf16*)alloc(256 * 256 * 2);
  float*  BIAS4 = (float*)alloc(NG4 * 4);
  float*  P1    = (float*)alloc((size_t)NBS * 256 * 4);
  float*  P2    = (float*)alloc((size_t)NBS * 256 * 4);
  float*  X     = (float*)alloc((size_t)NBS * NG4 * 4);
  float*  OLFa  = (float*)alloc((size_t)NBS * 256 * 4);
  float*  OLFb  = (float*)alloc((size_t)NBS * 256 * 4);
  float*  H     = (float*)alloc((size_t)NB * 256 * 4);
  float*  Cst   = (float*)alloc((size_t)NB * 256 * 4);
  float*  Z     = (float*)alloc((size_t)NB * NG4 * 4);
  float*  GS    = (float*)alloc((size_t)NBSS * 4);
  float*  BM    = (float*)alloc((size_t)NBS * 256 * 4);
  float*  SMS   = (float*)alloc((size_t)NBS * 256 * 4);
  float*  GM    = (float*)alloc((size_t)NBS * 4);
  float*  GSUM  = (float*)alloc((size_t)NBS * 4);

  prep_kernel<<<512, 256, 0, stream>>>(W_ih, W_hh, Wgr, Wbc, Wesa, b_ih, b_hh,
                                       WIH, WHH, W1, W2, W3, WBCb, WESb, BIAS4);
  olf_init_kernel<<<(unsigned)((size_t)NBS * 256 / 256), 256, 0, stream>>>(emb, counts, OLFa);
  rlf0_kernel<<<(unsigned)NBSS, 256, 0, stream>>>(centers, counts, Wr, br, RLF);

  float* olf_cur = OLFa;
  float* olf_nxt = OLFb;
  for (int L = 0; L < 3; ++L) {
    // X = olf @ W_ih^T + (b_ih + b_hh)
    dim3 gX(NBS / 32, NG4 / 64);
    gemm_bf16_kernel<<<gX, 256, 0, stream>>>(olf_cur, 256, WIH, 256, BIAS4,
                                             nullptr, 0, X, NG4, NBS);
    zero_kernel<<<(NB * 256 + 255) / 256, 256, 0, stream>>>(H, NB * 256);
    zero_kernel<<<(NB * 256 + 255) / 256, 256, 0, stream>>>(Cst, NB * 256);
    for (int t = 0; t < NS; ++t) {
      dim3 gZ(NB / 32, NG4 / 64);
      gemm_bf16_kernel<<<gZ, 256, 0, stream>>>(H, 256, WHH, 256, nullptr,
                                               X + (size_t)t * NG4, (long long)NS * NG4,
                                               Z, NG4, NB);
      lstm_gate_kernel<<<(NB * NO + 255) / 256, 256, 0, stream>>>(Z, H, Cst, olf_nxt, t, counts);
    }
    // P1 = olf@W1^T, P2 = olf@W2^T (bgr folded into fused kernel)
    dim3 gP(NBS / 32, 256 / 64);
    gemm_bf16_kernel<<<gP, 256, 0, stream>>>(olf_cur, 256, W1, 256, nullptr, nullptr, 0, P1, 256, NBS);
    gemm_bf16_kernel<<<gP, 256, 0, stream>>>(olf_cur, 256, W2, 256, nullptr, nullptr, 0, P2, 256, NBS);
    // fused relation update (in place on RLF), 64 rows per 512-thread block
    layer_fused_kernel<<<(unsigned)(NBSS / 64), 512, 0, stream>>>(RLF, P1, P2, W3, bgr,
                                                                  WBCb, bbc, alpha, counts);
    float* tmp = olf_cur; olf_cur = olf_nxt; olf_nxt = tmp;
  }

  // R2 = RLF @ Wesa^T + besa
  dim3 gE((unsigned)(NBSS / 32), 256 / 64);
  gemm_bf16_kernel<<<gE, 256, 0, stream>>>(RLF, 256, WESb, 256, besa, nullptr, 0, R2, 256,
                                           (int)NBSS);
  gscore_kernel<<<(unsigned)(NBSS / 8), 256, 0, stream>>>(R2, Wmu, bmu, GS);
  softmax_stats_kernel<<<NBS, 256, 0, stream>>>(R2, GS, BM, SMS, GM, GSUM);
  attn_out_kernel<<<NBS, 256, 0, stream>>>(R2, GS, BM, SMS, GM, GSUM, counts, (float*)d_out);
}
